// Attention_11158325035104
// MI455X (gfx1250) — matrix operand order repaired and checked by reading, not compiled, not run on hardware
//
#include <hip/hip_runtime.h>

typedef __attribute__((ext_vector_type(16))) __bf16 v16bf;
typedef __attribute__((ext_vector_type(8)))  float  v8f;
typedef unsigned short u16w;

#define T_ENC 64
#define D_ENC 256
#define H_DIM 256
#define C_CLS 96
#define H4    1024

__device__ __forceinline__ float fexp2(float x){ return __builtin_amdgcn_exp2f(x); }
__device__ __forceinline__ float frcp (float x){ return __builtin_amdgcn_rcpf(x); }

#if __has_builtin(__builtin_amdgcn_tanhf)
// gfx1250 native v_tanh_f32
__device__ __forceinline__ float fast_tanh(float x){ return __builtin_amdgcn_tanhf(x); }
__device__ __forceinline__ float fast_sigmoid(float x){
  return __builtin_fmaf(0.5f, __builtin_amdgcn_tanhf(0.5f * x), 0.5f);
}
#else
// fallback: tanh(x) = 1 - 2/(exp2(2x*log2e)+1)
__device__ __forceinline__ float fast_tanh(float x){
  return 1.0f - 2.0f * frcp(fexp2(x * 2.8853900817779268f) + 1.0f);
}
__device__ __forceinline__ float fast_sigmoid(float x){
  return frcp(1.0f + fexp2(x * -1.4426950408889634f));
}
#endif

// f32 -> bf16 via native convert (v_cvt_*bf16_f32 on gfx1250)
__device__ __forceinline__ __bf16 f2bf(float f){ return (__bf16)f; }
__device__ __forceinline__ u16w f2bfbits(float f){
  __bf16 b = (__bf16)f;
  return __builtin_bit_cast(u16w, b);
}

// -------------------------------------------------------------------------
// Pack f32 KxN row-major weight into WMMA B-fragment order (bf16).
// Fragment (kblk,nblk): 32 lanes x 16 halfs, contiguous per lane.
//   e<8 : K = kblk*32 + 8*(lane/16) + e      e>=8 : K = kblk*32 + 16 + 8*(lane/16) + (e-8)
//   col = nblk*16 + lane%16
// -------------------------------------------------------------------------
__global__ void pack_b_kernel(const float* __restrict__ W, u16w* __restrict__ out,
                              int K, int N){
  int total = (K >> 5) * (N >> 4) * 512;
  int nblk_count = N >> 4;
  for (int idx = blockIdx.x * blockDim.x + threadIdx.x; idx < total;
       idx += gridDim.x * blockDim.x){
    int f = idx >> 9, r = idx & 511;
    int lane = r >> 4, e = r & 15;
    int kblk = f / nblk_count, nblk = f - kblk * nblk_count;
    int k = kblk * 32 + ((lane >> 4) << 3) + (e & 7) + ((e >> 3) << 4);
    int col = nblk * 16 + (lane & 15);
    out[idx] = f2bfbits(W[(size_t)k * N + col]);
  }
}

// -------------------------------------------------------------------------
// proj = batch_H @ Wi2h   (rows = B*T, K = 256, N = 256), one-time GEMM.
// 8 waves/block, each wave owns one 16x16 output tile.
// -------------------------------------------------------------------------
__global__ void __launch_bounds__(256) proj_kernel(const float* __restrict__ bH,
                                                   const u16w* __restrict__ wi2h_p,
                                                   float* __restrict__ proj, int rows){
  int lane = threadIdx.x & 31, wv = threadIdx.x >> 5;
  int gw = blockIdx.x * 8 + wv;
  int rt = gw >> 4, ct = gw & 15;
  if (rt * 16 >= rows) return;
  const float* arow = bH + (size_t)(rt * 16 + (lane & 15)) * D_ENC;
  int khi = (lane >> 4) << 3;   // A-fragment: lane<16 -> K base 0, lane>=16 -> +8
  v8f acc = {};
  #pragma unroll
  for (int kb = 0; kb < 8; ++kb){
    v16bf a;
    #pragma unroll
    for (int e = 0; e < 8; ++e) a[e]     = f2bf(arow[kb*32 + khi + e]);
    #pragma unroll
    for (int e = 0; e < 8; ++e) a[8 + e] = f2bf(arow[kb*32 + khi + 16 + e]);
    v16bf b = *reinterpret_cast<const v16bf*>(
        &wi2h_p[((size_t)(kb * 16 + ct) * 32 + lane) * 16]);
    acc = __builtin_amdgcn_wmma_f32_16x16x32_bf16(false, a, false, b, (short)0, acc,
                                                  false, false);
  }
  int m0 = (lane >> 4) << 3;
  int col = ct * 16 + (lane & 15);
  #pragma unroll
  for (int v = 0; v < 8; ++v)
    proj[(size_t)(rt * 16 + m0 + v) * H_DIM + col] = acc[v];
}

// -------------------------------------------------------------------------
// Fused 26-step attention-LSTM scan. 1 block = 16 batch rows, 8 waves.
// All per-step GEMMs via v_wmma_f32_16x16x32_bf16; everything L2-resident.
// -------------------------------------------------------------------------
__global__ void __launch_bounds__(256) attn_scan_kernel(
    const float* __restrict__ batch_H,   // [B,64,256] f32
    const int*   __restrict__ text,      // [B,num_steps] i32
    const float* __restrict__ Wx,        // [352,1024] f32 (one-hot rows 256..351)
    const float* __restrict__ b_lstm,    // [1024]
    const float* __restrict__ bh2h,      // [256]
    const float* __restrict__ w_score,   // [256]
    const float* __restrict__ bg,        // [96]
    const float* __restrict__ proj,      // [B,64,256] f32 (precomputed)
    const u16w* __restrict__ wh2h_p, const u16w* __restrict__ wx_p,
    const u16w* __restrict__ wh_p,   const u16w* __restrict__ wg_p,
    float* __restrict__ out,             // [B,num_steps,96] f32
    int num_steps)
{
  __shared__ __attribute__((aligned(32))) u16w h_pack[8*32*16];    // h, A-frag bf16
  __shared__ __attribute__((aligned(32))) u16w ctx_pack[8*32*16];  // ctx, A-frag bf16
  __shared__ __attribute__((aligned(16))) float hp_s[16*H_DIM];
  __shared__ __attribute__((aligned(16))) float c_s[16*H_DIM];
  __shared__ __attribute__((aligned(16))) float e_s[16*T_ENC];
  __shared__ __attribute__((aligned(16))) float ws_s[H_DIM];
  __shared__ int chars_s[16];

  const int tid  = threadIdx.x;
  const int lane = tid & 31;
  const int wv   = tid >> 5;            // wave 0..7
  const int b0   = blockIdx.x * 16;

  for (int i = tid; i < 8*32*16;  i += 256) h_pack[i] = 0;
  for (int i = tid; i < 16*H_DIM; i += 256) c_s[i] = 0.0f;
  if (tid < H_DIM) ws_s[tid] = w_score[tid];
  __syncthreads();

#define LDA(buf, kb) (*reinterpret_cast<const v16bf*>(&(buf)[((kb)*32 + lane)*16]))
#define LDB(ptr, kb, ntiles, nt) \
  (*reinterpret_cast<const v16bf*>(&(ptr)[((size_t)((kb)*(ntiles) + (nt))*32 + lane)*16]))

  for (int step = 0; step < num_steps; ++step){
    if (tid < 16) chars_s[tid] = text[(size_t)(b0 + tid) * num_steps + step];

    // ---- phase 1: hp = h @ Wh2h + bh2h  (16x256, K=256; 2 tiles/wave) ----
    #pragma unroll
    for (int j = 0; j < 2; ++j){
      int nt = wv * 2 + j;
      v8f acc = {};
      #pragma unroll
      for (int kb = 0; kb < 8; ++kb){
        v16bf a = LDA(h_pack, kb);
        v16bf b = LDB(wh2h_p, kb, 16, nt);
        acc = __builtin_amdgcn_wmma_f32_16x16x32_bf16(false, a, false, b, (short)0, acc,
                                                      false, false);
      }
      int m0  = (lane >> 4) << 3;
      int col = nt * 16 + (lane & 15);
      float bias = bh2h[col];
      #pragma unroll
      for (int v = 0; v < 8; ++v) hp_s[(m0 + v) * H_DIM + col] = acc[v] + bias;
    }
    __syncthreads();

    // ---- phase 2: e[b][t] = sum_h tanh(proj+hp) * w_score (v_tanh_f32) ----
    #pragma unroll
    for (int i = 0; i < 4; ++i){
      int p = tid + 256 * i;                 // 1024 (b,t) pairs
      int b = p >> 6, t = p & 63;
      const float* pr  = proj + ((size_t)(b0 + b) * T_ENC + t) * H_DIM;
      const float* hpr = &hp_s[b * H_DIM];
      float sum = 0.0f;
      for (int h = 0; h < H_DIM; h += 4){
        float4 pv  = *reinterpret_cast<const float4*>(pr + h);
        float4 hv  = *reinterpret_cast<const float4*>(hpr + h);
        float4 wsv = *reinterpret_cast<const float4*>(&ws_s[h]);
        sum += fast_tanh(pv.x + hv.x) * wsv.x;
        sum += fast_tanh(pv.y + hv.y) * wsv.y;
        sum += fast_tanh(pv.z + hv.z) * wsv.z;
        sum += fast_tanh(pv.w + hv.w) * wsv.w;
      }
      e_s[p] = sum;
    }
    __syncthreads();

    // ---- softmax over T per row ----
    if (tid < 16){
      float mx = -3.4e38f;
      for (int t = 0; t < T_ENC; ++t) mx = fmaxf(mx, e_s[tid * T_ENC + t]);
      float sum = 0.0f;
      for (int t = 0; t < T_ENC; ++t){
        float ex = fexp2((e_s[tid * T_ENC + t] - mx) * 1.4426950408889634f);
        e_s[tid * T_ENC + t] = ex;
        sum += ex;
      }
      float inv = frcp(sum);
      for (int t = 0; t < T_ENC; ++t) e_s[tid * T_ENC + t] *= inv;
    }
    __syncthreads();

    // ---- phase 3: context = sum_t alpha * batch_H -> ctx_pack (A-frag bf16) ----
    #pragma unroll
    for (int i = 0; i < 4; ++i){
      int task = tid + 256 * i;              // 16 rows x 64 float4 chunks
      int b = task >> 6;
      int d = (task & 63) * 4;
      const float* bh = batch_H + (size_t)(b0 + b) * T_ENC * D_ENC + d;
      float sx = 0, sy = 0, sz = 0, sw = 0;
      for (int t = 0; t < T_ENC; ++t){
        float al = e_s[b * T_ENC + t];
        float4 xv = *reinterpret_cast<const float4*>(bh + t * D_ENC);
        sx += al * xv.x; sy += al * xv.y; sz += al * xv.z; sw += al * xv.w;
      }
      int koff = d & 31, kblk = d >> 5;
      int lane2, pos;
      if      (koff <  8){ lane2 = b;      pos = koff;      }
      else if (koff < 16){ lane2 = 16 + b; pos = koff - 8;  }
      else if (koff < 24){ lane2 = b;      pos = koff - 8;  }
      else               { lane2 = 16 + b; pos = koff - 16; }
      u16w* dst = &ctx_pack[(kblk * 32 + lane2) * 16 + pos];
      dst[0] = f2bfbits(sx); dst[1] = f2bfbits(sy);
      dst[2] = f2bfbits(sz); dst[3] = f2bfbits(sw);
    }
    __syncthreads();

    // ---- phase 4: gates = ctx@Wx[0:256] + h@Wh  (16x1024, K=256+256) ----
    // wave wv owns gate tiles nt = 16*g + wv + 8*j2  -> holds i,f,c,o for same cols
    v8f acc[4][2];
    #pragma unroll
    for (int g = 0; g < 4; ++g)
      #pragma unroll
      for (int j2 = 0; j2 < 2; ++j2){ v8f z = {}; acc[g][j2] = z; }
    #pragma unroll
    for (int kb = 0; kb < 8; ++kb){
      v16bf a = LDA(ctx_pack, kb);
      #pragma unroll
      for (int g = 0; g < 4; ++g)
        #pragma unroll
        for (int j2 = 0; j2 < 2; ++j2){
          v16bf b = LDB(wx_p, kb, 64, 16*g + wv + 8*j2);
          acc[g][j2] = __builtin_amdgcn_wmma_f32_16x16x32_bf16(false, a, false, b,
                          (short)0, acc[g][j2], false, false);
        }
    }
    #pragma unroll
    for (int kb = 0; kb < 8; ++kb){
      v16bf a = LDA(h_pack, kb);
      #pragma unroll
      for (int g = 0; g < 4; ++g)
        #pragma unroll
        for (int j2 = 0; j2 < 2; ++j2){
          v16bf b = LDB(wh_p, kb, 64, 16*g + wv + 8*j2);
          acc[g][j2] = __builtin_amdgcn_wmma_f32_16x16x32_bf16(false, a, false, b,
                          (short)0, acc[g][j2], false, false);
        }
    }
    __syncthreads();   // all reads of old h_pack complete before rewrite

    // ---- LSTM elementwise: + one-hot Wx row + bias; update c, write new h ----
    #pragma unroll
    for (int j2 = 0; j2 < 2; ++j2){
      int u = (wv + 8*j2) * 16 + (lane & 15);        // hidden unit 0..255
      float bi = b_lstm[u], bfv = b_lstm[256+u], bc = b_lstm[512+u], bo = b_lstm[768+u];
      #pragma unroll
      for (int v = 0; v < 8; ++v){
        int m = ((lane >> 4) << 3) + v;              // batch row in tile
        const float* wrow = Wx + (size_t)(D_ENC + chars_s[m]) * H4;  // one-hot row
        float iv = fast_sigmoid(acc[0][j2][v] + bi  + wrow[u]);
        float fv = fast_sigmoid(acc[1][j2][v] + bfv + wrow[256+u]);
        float gv = fast_tanh   (acc[2][j2][v] + bc  + wrow[512+u]);
        float ov = fast_sigmoid(acc[3][j2][v] + bo  + wrow[768+u]);
        float cn = fv * c_s[m * H_DIM + u] + iv * gv;
        c_s[m * H_DIM + u] = cn;
        float hn = ov * fast_tanh(cn);
        int koff = u & 31, kblk = u >> 5;
        int lane2, pos;
        if      (koff <  8){ lane2 = m;      pos = koff;      }
        else if (koff < 16){ lane2 = 16 + m; pos = koff - 8;  }
        else if (koff < 24){ lane2 = m;      pos = koff - 8;  }
        else               { lane2 = 16 + m; pos = koff - 16; }
        h_pack[(kblk * 32 + lane2) * 16 + pos] = f2bfbits(hn);
      }
    }
    __syncthreads();

    // ---- phase 5 (fused generator): probs[:,step,:] = h_new @ Wg + bg ----
    if (wv < 6){                                     // 6 N-tiles of 16 = 96 classes
      v8f acc2 = {};
      #pragma unroll
      for (int kb = 0; kb < 8; ++kb){
        v16bf a = LDA(h_pack, kb);
        v16bf b = LDB(wg_p, kb, 6, wv);
        acc2 = __builtin_amdgcn_wmma_f32_16x16x32_bf16(false, a, false, b, (short)0,
                                                       acc2, false, false);
      }
      int m0  = (lane >> 4) << 3;
      int col = wv * 16 + (lane & 15);
      float bias = bg[col];
      #pragma unroll
      for (int v = 0; v < 8; ++v)
        out[((size_t)(b0 + m0 + v) * num_steps + step) * C_CLS + col] = acc2[v] + bias;
    }
    __syncthreads();
  }
#undef LDA
#undef LDB
}

// -------------------------------------------------------------------------
extern "C" void kernel_launch(void* const* d_in, const int* in_sizes, int n_in,
                              void* d_out, int out_size, void* d_ws, size_t ws_size,
                              hipStream_t stream) {
  const float* batch_H = (const float*)d_in[0];
  const int*   text    = (const int*)d_in[1];
  const float* Wi2h    = (const float*)d_in[3];
  const float* Wh2h    = (const float*)d_in[4];
  const float* bh2h    = (const float*)d_in[5];
  const float* w_score = (const float*)d_in[6];
  const float* Wx      = (const float*)d_in[7];
  const float* Wh      = (const float*)d_in[8];
  const float* b_lstm  = (const float*)d_in[9];
  const float* Wg      = (const float*)d_in[10];
  const float* bg      = (const float*)d_in[11];

  const int B = in_sizes[0] / (T_ENC * D_ENC);       // 1024
  const int num_steps = in_sizes[1] / B;             // batch_max_length + 1 = 26

  // workspace layout (all chunk sizes multiples of 256B -> alignment preserved)
  char* ws = (char*)d_ws;
  size_t o = 0;
  u16w* wi2h_p = (u16w*)(ws + o); o += (size_t)D_ENC * H_DIM * 2;   // 128 KB
  u16w* wh2h_p = (u16w*)(ws + o); o += (size_t)H_DIM * H_DIM * 2;   // 128 KB
  u16w* wx_p   = (u16w*)(ws + o); o += (size_t)D_ENC * H4 * 2;      // 512 KB
  u16w* wh_p   = (u16w*)(ws + o); o += (size_t)H_DIM * H4 * 2;      // 512 KB
  u16w* wg_p   = (u16w*)(ws + o); o += (size_t)H_DIM * C_CLS * 2;   //  48 KB
  float* proj  = (float*)(ws + o); o += (size_t)B * T_ENC * H_DIM * 4; // 64 MB

  // 1) pack weights into bf16 B-fragment layout (every call; ws is not persistent)
  pack_b_kernel<<<256, 256, 0, stream>>>(Wi2h, wi2h_p, D_ENC, H_DIM);
  pack_b_kernel<<<256, 256, 0, stream>>>(Wh2h, wh2h_p, H_DIM, H_DIM);
  pack_b_kernel<<<512, 256, 0, stream>>>(Wx,   wx_p,   D_ENC, H4);  // context rows only
  pack_b_kernel<<<512, 256, 0, stream>>>(Wh,   wh_p,   H_DIM, H4);
  pack_b_kernel<<<64,  256, 0, stream>>>(Wg,   wg_p,   H_DIM, C_CLS);

  // 2) one-time encoder projection GEMM (result stays L2-resident)
  const int rows = B * T_ENC;
  proj_kernel<<<rows / 8, 256, 0, stream>>>(batch_H, wi2h_p, proj, rows);

  // 3) fused 26-step scan + generator
  attn_scan_kernel<<<B / 16, 256, 0, stream>>>(batch_H, text, Wx, b_lstm, bh2h,
      w_score, bg, proj, wh2h_p, wx_p, wh_p, wg_p, (float*)d_out, num_steps);
}
